// Block_6442450944223
// MI455X (gfx1250) — compile-verified, matrix operand order checked by reading, not run on hardware
//
#include <hip/hip_runtime.h>

typedef unsigned short ushort_t;
typedef unsigned int uint_t;

typedef __attribute__((ext_vector_type(16))) __bf16          v16bf;
typedef __attribute__((ext_vector_type(16))) unsigned short  v16u;
typedef __attribute__((ext_vector_type(8)))  float           v8f;
typedef __attribute__((ext_vector_type(4)))  unsigned int    v4u;

#define DIM   768
#define NTOK  8192      // 8 * 1024
#define SEQ   1024
#define NQKV  2304
#define DFF   3072
#define HEADS 8
#define DH    96
#define LN_EPS 1e-5f

// ---------- bf16 helpers (bit-level, round-to-nearest-even) ----------
__device__ __forceinline__ ushort_t f2bf(float f) {
    uint_t u = __float_as_uint(f);
    u += 0x7FFFu + ((u >> 16) & 1u);
    return (ushort_t)(u >> 16);
}

// ---------- WMMA fragment: two contiguous 16B halves at lo and lo+16 ----------
// Matches CDNA5 16-bit A-layout: lane l<16 holds K {kb+0..7, kb+16..23} with kb=0,
// lanes 16..31 with kb=8.  Works for LDS and global pointers (16B aligned).
__device__ __forceinline__ v16u frag_pair(const ushort_t* lo) {
    union { v4u q[2]; v16u v; } u;
    u.q[0] = *(const v4u*)lo;
    u.q[1] = *(const v4u*)(lo + 16);
    return u.v;
}
__device__ __forceinline__ v8f wmma_bf16(v16u a, v16u b, v8f c) {
    return __builtin_amdgcn_wmma_f32_16x16x32_bf16(
        false, __builtin_bit_cast(v16bf, a),
        false, __builtin_bit_cast(v16bf, b),
        (short)0, c, false, false);
}

// ---------- fp32 -> bf16 conversion ----------
__global__ void k_f32_to_bf16(const float* __restrict__ in, ushort_t* __restrict__ out, int n) {
    for (int i = blockIdx.x * blockDim.x + threadIdx.x; i < n; i += gridDim.x * blockDim.x)
        out[i] = f2bf(in[i]);
}

// ---------- LayerNorm: one 256-thread block per 768-wide row, bf16 out ----------
__global__ __launch_bounds__(256) void k_layernorm_bf16(
    const float* __restrict__ x, const float* __restrict__ g, const float* __restrict__ b,
    ushort_t* __restrict__ out)
{
    __shared__ float red[256];
    const int row = blockIdx.x;
    const int t = threadIdx.x;
    const float* xr = x + (size_t)row * DIM;

    float v0 = xr[t], v1 = xr[t + 256], v2 = xr[t + 512];

    red[t] = v0 + v1 + v2;
    __syncthreads();
#pragma unroll
    for (int off = 128; off > 0; off >>= 1) {
        if (t < off) red[t] += red[t + off];
        __syncthreads();
    }
    const float mu = red[0] * (1.0f / (float)DIM);
    __syncthreads();

    const float d0 = v0 - mu, d1 = v1 - mu, d2 = v2 - mu;
    red[t] = d0 * d0 + d1 * d1 + d2 * d2;
    __syncthreads();
#pragma unroll
    for (int off = 128; off > 0; off >>= 1) {
        if (t < off) red[t] += red[t + off];
        __syncthreads();
    }
    const float var = red[0] * (1.0f / (float)DIM);
    const float rs = rsqrtf(var + LN_EPS);

    ushort_t* orow = out + (size_t)row * DIM;
    orow[t]       = f2bf(d0 * rs * g[t]       + b[t]);
    orow[t + 256] = f2bf(d1 * rs * g[t + 256] + b[t + 256]);
    orow[t + 512] = f2bf(d2 * rs * g[t + 512] + b[t + 512]);
}

// ---------- Tiled GEMM: C[M,N] = A[M,K](bf16) @ B[K,N](bf16) + bias (+gelu) (+residual) ----------
// 128x128 block tile, K staged 32 deep, double-buffered LDS (one barrier per stage).
// B tile stored TRANSPOSED in LDS ([n][k], ld=32) so every fragment read is 2x ds_load_b128.
__global__ __launch_bounds__(256) void k_gemm_bf16(
    const ushort_t* __restrict__ A, int lda,
    const ushort_t* __restrict__ B, int ldb,
    const float* __restrict__ bias,
    const float* __restrict__ res,
    float* __restrict__ outF, ushort_t* __restrict__ outB,
    int N, int K, int do_gelu)
{
    __shared__ ushort_t As[2][128 * 32];   // [row][k]
    __shared__ ushort_t Bs[2][128 * 32];   // [n][k]  (transposed)

    const int tid  = threadIdx.x;
    const int lane = tid & 31;
    const int wave = tid >> 5;
    const int wm = (wave & 1) * 64;
    const int wn = (wave >> 1) * 32;
    const int bm = blockIdx.y * 128;
    const int bn = blockIdx.x * 128;
    const int lrow = lane & 15;
    const int kb = (lane & 16) ? 8 : 0;

    // staging indices
    const int ar = tid >> 1, ac = (tid & 1) * 16;        // A: 128 rows x 32 k
    const int bk = tid >> 3, bc = (tid & 7) * 16;        // B: 32 k x 128 n

    const int nk = K >> 5;

    v8f acc[4][2];
#pragma unroll
    for (int i = 0; i < 4; ++i)
#pragma unroll
        for (int j = 0; j < 2; ++j) acc[i][j] = (v8f)0.0f;

    v4u aS0, aS1;                          // A stage regs (16 bf16)
    union { v4u q[2]; ushort_t u[16]; } bS; // B stage regs (16 bf16)

    auto load_stage = [&](int k0) {
        const ushort_t* ga = A + (size_t)(bm + ar) * lda + k0 + ac;
        aS0 = *(const v4u*)ga;
        aS1 = *(const v4u*)(ga + 8);
        const ushort_t* gb = B + (size_t)(k0 + bk) * ldb + bn + bc;
        bS.q[0] = *(const v4u*)gb;
        bS.q[1] = *(const v4u*)(gb + 8);
    };
    auto store_stage = [&](int buf) {
        *(v4u*)(&As[buf][ar * 32 + ac])     = aS0;
        *(v4u*)(&As[buf][ar * 32 + ac + 8]) = aS1;
#pragma unroll
        for (int i = 0; i < 16; ++i) Bs[buf][(bc + i) * 32 + bk] = bS.u[i];
    };

    load_stage(0);
    store_stage(0);
    __syncthreads();

    for (int s = 0; s < nk; ++s) {
        const int cur = s & 1;
        if (s + 1 < nk) load_stage((s + 1) << 5);   // prefetch next stage into regs

        v16u af[4], bf[2];
#pragma unroll
        for (int i = 0; i < 4; ++i)
            af[i] = frag_pair(&As[cur][(wm + i * 16 + lrow) * 32 + kb]);
#pragma unroll
        for (int j = 0; j < 2; ++j)
            bf[j] = frag_pair(&Bs[cur][(wn + j * 16 + lrow) * 32 + kb]);
#pragma unroll
        for (int i = 0; i < 4; ++i)
#pragma unroll
            for (int j = 0; j < 2; ++j) acc[i][j] = wmma_bf16(af[i], bf[j], acc[i][j]);

        if (s + 1 < nk) {
            store_stage(cur ^ 1);
            __syncthreads();
        }
    }

    const int rowAdd = (lane & 16) ? 8 : 0;
#pragma unroll
    for (int i = 0; i < 4; ++i) {
#pragma unroll
        for (int j = 0; j < 2; ++j) {
            const int gcol = bn + wn + j * 16 + lrow;
            const float bv = bias ? bias[gcol] : 0.0f;
#pragma unroll
            for (int r = 0; r < 8; ++r) {
                const int grow = bm + wm + i * 16 + r + rowAdd;
                float v = acc[i][j][r] + bv;
                if (do_gelu) v = 0.5f * v * (1.0f + erff(v * 0.70710678118654752f));
                if (res)  v += res[(size_t)grow * N + gcol];
                if (outF) outF[(size_t)grow * N + gcol] = v;
                if (outB) outB[(size_t)grow * N + gcol] = f2bf(v);
            }
        }
    }
}

// ---------- Flash attention: block = 64 query rows of one (b,h); 4 waves x 16 rows ----------
__global__ __launch_bounds__(128) void k_flash_attn(
    const ushort_t* __restrict__ qkv,   // [8192, 2304] bf16: [Q(768)|K(768)|V(768)], head-major 96 each
    ushort_t* __restrict__ o)           // [8192, 768] bf16
{
    __shared__ ushort_t Ks[32 * 96];        // [key][d]   (row-major, for S = Q K^T)
    __shared__ ushort_t Vt[96 * 32];        // [d][key]   (transposed, for O += P V)
    __shared__ ushort_t Ps[4][16 * 32];     // per-wave P tile, [row][key]

    const int bh = blockIdx.y;
    const int bb = bh >> 3, hh = bh & 7;
    const int wv = threadIdx.x >> 5;
    const int lane = threadIdx.x & 31;
    const int lrow = lane & 15;
    const int kbsel = (lane & 16) ? 8 : 0;
    const int rowAdd = (lane & 16) ? 8 : 0;
    const int r0 = blockIdx.x * 64 + wv * 16;   // query row tile within this batch

    const size_t tokbase = (size_t)bb * SEQ;
    const ushort_t* qb = qkv + tokbase * NQKV + hh * DH;
    const ushort_t* kp = qb + DIM;
    const ushort_t* vp = qb + 2 * DIM;

    // Q fragments resident in registers for the whole KV sweep (16 rows x 96)
    v16u qa[3];
#pragma unroll
    for (int kk = 0; kk < 3; ++kk)
        qa[kk] = frag_pair(qb + (size_t)(r0 + lrow) * NQKV + kk * 32 + kbsel);

    v8f oacc[6];
#pragma unroll
    for (int t = 0; t < 6; ++t) oacc[t] = (v8f)0.0f;
    float mrow[8], lsum[8];
#pragma unroll
    for (int r = 0; r < 8; ++r) { mrow[r] = -1e30f; lsum[r] = 0.0f; }

    const float scale = 0.10206207261596575f;   // 1/sqrt(96)

    // staging split: 128 threads, 32 keys x 96 d; thread -> key = t>>2, d = (t&3)*24
    const int skey = threadIdx.x >> 2;
    const int sd   = (threadIdx.x & 3) * 24;

    for (int j0 = 0; j0 < SEQ; j0 += 32) {
        {   // K row-major (3x b128 in, 3x b128 out); V transposed (3x b128 in, 24 b16 scatter)
            const ushort_t* gk = kp + (size_t)(j0 + skey) * NQKV + sd;
            const ushort_t* gv = vp + (size_t)(j0 + skey) * NQKV + sd;
            v4u k0v = *(const v4u*)gk, k1v = *(const v4u*)(gk + 8), k2v = *(const v4u*)(gk + 16);
            union { v4u q[3]; ushort_t u[24]; } vs;
            vs.q[0] = *(const v4u*)gv; vs.q[1] = *(const v4u*)(gv + 8); vs.q[2] = *(const v4u*)(gv + 16);
            *(v4u*)(&Ks[skey * 96 + sd])      = k0v;
            *(v4u*)(&Ks[skey * 96 + sd + 8])  = k1v;
            *(v4u*)(&Ks[skey * 96 + sd + 16]) = k2v;
#pragma unroll
            for (int i = 0; i < 24; ++i) Vt[(sd + i) * 32 + skey] = vs.u[i];
        }
        __syncthreads();

        // S = Q @ K^T : two 16-wide key tiles, d-dim = 96 (3 WMMA K-steps each)
        v8f s0 = (v8f)0.0f, s1 = (v8f)0.0f;
#pragma unroll
        for (int kk = 0; kk < 3; ++kk) {
            v16u b0 = frag_pair(&Ks[lrow * 96 + kk * 32 + kbsel]);          // keys j0+0..15
            v16u b1 = frag_pair(&Ks[(16 + lrow) * 96 + kk * 32 + kbsel]);   // keys j0+16..31
            s0 = wmma_bf16(qa[kk], b0, s0);
            s1 = wmma_bf16(qa[kk], b1, s1);
        }

        // online softmax (per row: 16-lane group reductions via xor shuffles)
#pragma unroll
        for (int r = 0; r < 8; ++r) {
            float a0 = s0[r] * scale, a1 = s1[r] * scale;
            float mx = fmaxf(a0, a1);
#pragma unroll
            for (int off = 1; off <= 8; off <<= 1) mx = fmaxf(mx, __shfl_xor(mx, off, 32));
            const float mnew = fmaxf(mrow[r], mx);
            const float alpha = __expf(mrow[r] - mnew);
            mrow[r] = mnew;
            const float p0 = __expf(a0 - mnew);
            const float p1 = __expf(a1 - mnew);
            float ps = p0 + p1;
#pragma unroll
            for (int off = 1; off <= 8; off <<= 1) ps += __shfl_xor(ps, off, 32);
            lsum[r] = lsum[r] * alpha + ps;
#pragma unroll
            for (int t = 0; t < 6; ++t) oacc[t][r] *= alpha;
            const int prow = r + rowAdd;
            Ps[wv][prow * 32 + lrow]      = f2bf(p0);
            Ps[wv][prow * 32 + 16 + lrow] = f2bf(p1);
        }
        __syncthreads();

        // O += P @ V : key-dim = 32, 6 output tiles across Dh=96
        v16u pa = frag_pair(&Ps[wv][lrow * 32 + kbsel]);
#pragma unroll
        for (int t = 0; t < 6; ++t) {
            v16u vf = frag_pair(&Vt[(t * 16 + lrow) * 32 + kbsel]);
            oacc[t] = wmma_bf16(pa, vf, oacc[t]);
        }
        __syncthreads();
    }

    // normalize and store
#pragma unroll
    for (int t = 0; t < 6; ++t) {
#pragma unroll
        for (int r = 0; r < 8; ++r) {
            const int grow = r0 + r + rowAdd;
            const int gcol = hh * DH + t * 16 + lrow;
            o[(tokbase + grow) * DIM + gcol] = f2bf(oacc[t][r] / lsum[r]);
        }
    }
}

// ---------------------------------------------------------------------------
extern "C" void kernel_launch(void* const* d_in, const int* in_sizes, int n_in,
                              void* d_out, int out_size, void* d_ws, size_t ws_size,
                              hipStream_t stream) {
    (void)in_sizes; (void)n_in; (void)out_size; (void)ws_size;

    const float* x      = (const float*)d_in[0];
    const float* ln1_g  = (const float*)d_in[1];
    const float* ln1_b  = (const float*)d_in[2];
    const float* w_qkv  = (const float*)d_in[3];
    const float* w_proj = (const float*)d_in[4];
    const float* b_proj = (const float*)d_in[5];
    const float* ln2_g  = (const float*)d_in[6];
    const float* ln2_b  = (const float*)d_in[7];
    const float* w_fc1  = (const float*)d_in[8];
    const float* b_fc1  = (const float*)d_in[9];
    const float* w_fc2  = (const float*)d_in[10];
    const float* b_fc2  = (const float*)d_in[11];
    float* out = (float*)d_out;

    // workspace carve-up (256B aligned)
    size_t off = 0;
    auto carve = [&](size_t bytes) -> char* {
        char* p = (char*)d_ws + off;
        off += (bytes + 255) & ~(size_t)255;
        return p;
    };
    ushort_t* wqkv_b  = (ushort_t*)carve((size_t)DIM * NQKV * 2);
    ushort_t* wproj_b = (ushort_t*)carve((size_t)DIM * DIM  * 2);
    ushort_t* wfc1_b  = (ushort_t*)carve((size_t)DIM * DFF  * 2);
    ushort_t* wfc2_b  = (ushort_t*)carve((size_t)DFF * DIM  * 2);
    ushort_t* hb      = (ushort_t*)carve((size_t)NTOK * DIM  * 2);  // LN1 out
    ushort_t* qkv_b   = (ushort_t*)carve((size_t)NTOK * NQKV * 2);  // QKV out
    ushort_t* ob      = (ushort_t*)carve((size_t)NTOK * DIM  * 2);  // attn out
    float*    x1      = (float*)   carve((size_t)NTOK * DIM  * 4);  // post-proj residual
    ushort_t* h2b     = (ushort_t*)carve((size_t)NTOK * DIM  * 2);  // LN2 out
    ushort_t* h3b     = (ushort_t*)carve((size_t)NTOK * DFF  * 2);  // FC1+GELU out

    // 1) weights -> bf16
    auto cvt = [&](const float* src, ushort_t* dst, int n) {
        int blocks = (n + 255) / 256; if (blocks > 8192) blocks = 8192;
        k_f32_to_bf16<<<blocks, 256, 0, stream>>>(src, dst, n);
    };
    cvt(w_qkv,  wqkv_b,  DIM * NQKV);
    cvt(w_proj, wproj_b, DIM * DIM);
    cvt(w_fc1,  wfc1_b,  DIM * DFF);
    cvt(w_fc2,  wfc2_b,  DFF * DIM);

    // 2) LN1
    k_layernorm_bf16<<<NTOK, 256, 0, stream>>>(x, ln1_g, ln1_b, hb);

    // 3) QKV GEMM: [8192,768] @ [768,2304] -> bf16
    k_gemm_bf16<<<dim3(NQKV / 128, NTOK / 128), 256, 0, stream>>>(
        hb, DIM, wqkv_b, NQKV, nullptr, nullptr, nullptr, qkv_b, NQKV, DIM, 0);

    // 4) flash attention over 64 (b,h) pairs
    k_flash_attn<<<dim3(SEQ / 64, 64), 128, 0, stream>>>(qkv_b, ob);

    // 5) proj GEMM + bias + residual(x) -> x1 (fp32)
    k_gemm_bf16<<<dim3(DIM / 128, NTOK / 128), 256, 0, stream>>>(
        ob, DIM, wproj_b, DIM, b_proj, x, x1, nullptr, DIM, DIM, 0);

    // 6) LN2
    k_layernorm_bf16<<<NTOK, 256, 0, stream>>>(x1, ln2_g, ln2_b, h2b);

    // 7) FC1 + bias + GELU -> bf16
    k_gemm_bf16<<<dim3(DFF / 128, NTOK / 128), 256, 0, stream>>>(
        h2b, DIM, wfc1_b, DFF, b_fc1, nullptr, nullptr, h3b, DFF, DIM, 1);

    // 8) FC2 + bias + residual(x1) -> d_out (fp32)
    k_gemm_bf16<<<dim3(DIM / 128, NTOK / 128), 256, 0, stream>>>(
        h3b, DFF, wfc2_b, DIM, b_fc2, x1, out, nullptr, DIM, DFF, 0);
}
